// LSTMDecoder_19499151524332
// MI455X (gfx1250) — matrix operand order repaired and checked by reading, not compiled, not run on hardware
//
#include <hip/hip_runtime.h>
#include <hip/hip_bf16.h>

// ---------------------------------------------------------------------------
// LSTM decoder, fused for gfx1250 (MI455X).
//   B=512, LAT=64, H=1024, OUT=49, S=128, L=2
// Strategy:
//   * bf16 WMMA (v_wmma_f32_16x16x32_bf16) for every matmul, f32 accumulators.
//   * 1 workgroup = 16 batch rows; whole recurrence (both layers + output
//     head) runs inside one kernel, state resident in LDS.
//   * Weights pre-swizzled into B-fragment layout (pack kernels) so each lane
//     loads 32 contiguous bytes per fragment (2x global_load_b128 from L2).
// ---------------------------------------------------------------------------

typedef __bf16 bf16_t;
typedef __attribute__((ext_vector_type(16))) __bf16 v16bf;
typedef __attribute__((ext_vector_type(8)))  float  v8f;

#define HDIM    1024
#define KCH     32          // K chunks of 32 in HDIM
#define HPITCH  1034        // padded LDS row pitch (elements); 5-dword bank stride
#define BT      16          // batch tile (one WMMA M tile)
#define SLEN    128
#define NOUT    49
#define NTHREADS 512        // 16 waves
#define HTPW    4           // h-tiles per wave: 64 h-tiles / 16 waves

__device__ __forceinline__ float sigmoidf_(float x) {
  return 1.0f / (1.0f + __expf(-x));
}
__device__ __forceinline__ float tanhf_(float x) {
  float e = __expf(2.0f * x);
  return (e - 1.0f) / (e + 1.0f);
}

__device__ __forceinline__ v8f wmma_bf16(v16bf a, v16bf b, v8f c) {
  // 8 args: (neg_a, A, neg_b, B, c_mod, C, reuse_a, reuse_b)
  return __builtin_amdgcn_wmma_f32_16x16x32_bf16(false, a, false, b,
                                                 (short)0, c, false, false);
}

// A fragment (16x32 bf16, rows m = lane&15) gathered from an LDS row-major
// buffer, following the CDNA5 16-bit A layout:
//   lanes<16 : K = {k0+0..7, k0+16..23},  lanes>=16 : K = {k0+8..15, k0+24..31}
__device__ __forceinline__ v16bf load_afrag_lds(const bf16_t* hbuf, int k0, int lane) {
  int m    = lane & 15;
  int half = lane >> 4;
  const unsigned int* row = (const unsigned int*)(hbuf + m * HPITCH);
  int kb = (k0 + half * 8) >> 1;  // in 2-element (dword) units
  union { v16bf v; unsigned int u[8]; } r;
#pragma unroll
  for (int q = 0; q < 4; ++q) {
    r.u[q]     = row[kb + q];      // K = k0+half*8  + 2q,2q+1
    r.u[4 + q] = row[kb + 8 + q];  // K = k0+16+half*8 + 2q,2q+1
  }
  return r.v;
}

// B fragment: pre-swizzled, lane reads 16 contiguous bf16 (32 bytes).
__device__ __forceinline__ v16bf load_bfrag(const bf16_t* frag, int lane) {
  return ((const v16bf*)frag)[lane];
}

// Elementwise LSTM cell on the 4 gate accumulators (same frag mapping each):
// C-frag layout: VGPR v, lane l  ->  m = v + 8*(l>>4), n = ht*16 + (l&15)
__device__ __forceinline__ void cell_update(v8f acc[4], int ht, int lane,
                                            float* cbuf, bf16_t* hnext,
                                            const float* __restrict__ bA,
                                            const float* __restrict__ bB) {
  int n    = ht * 16 + (lane & 15);
  int half = lane >> 4;
  float bi = bA[n]            + bB[n];
  float bf = bA[HDIM + n]     + bB[HDIM + n];
  float bg = bA[2 * HDIM + n] + bB[2 * HDIM + n];
  float bo = bA[3 * HDIM + n] + bB[3 * HDIM + n];
#pragma unroll
  for (int v = 0; v < 8; ++v) {
    int m = v + half * 8;
    float i_ = sigmoidf_(acc[0][v] + bi);
    float f_ = sigmoidf_(acc[1][v] + bf);
    float g_ = tanhf_  (acc[2][v] + bg);
    float o_ = sigmoidf_(acc[3][v] + bo);
    float c  = f_ * cbuf[m * HPITCH + n] + i_ * g_;
    cbuf[m * HPITCH + n]  = c;
    hnext[m * HPITCH + n] = (bf16_t)(o_ * tanhf_(c));
  }
}

// ---------------------------------------------------------------------------
// Weight pack: W row-major [realN x K] f32  ->  bf16 B-fragments.
// Fragment (nt, kc): lane holds n = nt*16+(lane&15), K = kc*32+(lane>>4)*8+(j&7)+(j>>3)*16
// (j = 0..15, same k order as load_afrag_lds), flat index ((nt*kch + kc)*32 + lane)*16 + j.
// ---------------------------------------------------------------------------
__global__ void pack_weights(const float* __restrict__ W, bf16_t* __restrict__ out,
                             int ntiles, int realN, int kch, int K) {
  long long idx = (long long)blockIdx.x * blockDim.x + threadIdx.x;
  long long total = (long long)ntiles * kch * 512;
  if (idx >= total) return;
  int j    = (int)(idx & 15);
  int lane = (int)((idx >> 4) & 31);
  long long rest = idx >> 9;
  int kc = (int)(rest % kch);
  int nt = (int)(rest / kch);
  int n  = nt * 16 + (lane & 15);
  int k  = kc * 32 + (lane >> 4) * 8 + (j & 7) + (j >> 3) * 16;
  float v = (n < realN) ? W[(long long)n * K + k] : 0.0f;
  out[idx] = (bf16_t)v;
}

// ---------------------------------------------------------------------------
// Fused LSTM kernel: one workgroup per 16 batch rows.
// ---------------------------------------------------------------------------
__global__ void __launch_bounds__(NTHREADS, 1)
lstm_fused(const float* __restrict__ z,
           const float* __restrict__ W_hid, const float* __restrict__ b_hid,
           const float* __restrict__ W_cell, const float* __restrict__ b_cell,
           const float* __restrict__ b_ih0, const float* __restrict__ b_hh0,
           const float* __restrict__ b_ih1, const float* __restrict__ b_hh1,
           const float* __restrict__ b_out,
           const bf16_t* __restrict__ Bhh0, const bf16_t* __restrict__ Bih1,
           const bf16_t* __restrict__ Bhh1, const bf16_t* __restrict__ Bout,
           float* __restrict__ out) {
  extern __shared__ char smem[];
  const int HB = BT * HPITCH * (int)sizeof(bf16_t);   // 33088 B
  const int CB = BT * HPITCH * (int)sizeof(float);    // 66176 B
  bf16_t* h0a = (bf16_t*)(smem);
  bf16_t* h0b = (bf16_t*)(smem + HB);
  bf16_t* h1a = (bf16_t*)(smem + 2 * HB);
  bf16_t* h1b = (bf16_t*)(smem + 3 * HB);
  float*  c0  = (float*)(smem + 4 * HB);
  float*  c1  = (float*)(smem + 4 * HB + CB);

  const int tid   = threadIdx.x;
  const int lane  = tid & 31;
  const int wave  = tid >> 5;
  const int bbase = blockIdx.x * BT;

  // ---- init: h0/c0 per layer from z @ W_hid.T + b (K = 64, trivial) -------
  for (int idx = tid; idx < 2 * BT * HDIM; idx += NTHREADS) {
    int n = idx & (HDIM - 1);
    int m = (idx >> 10) & (BT - 1);
    int l = idx >> 14;
    int r = l * HDIM + n;
    const float* zr = z + (bbase + m) * 64;
    const float* wh = W_hid  + (long long)r * 64;
    const float* wc = W_cell + (long long)r * 64;
    float sh = b_hid[r], sc = b_cell[r];
    for (int i = 0; i < 64; ++i) {
      float zv = zr[i];
      sh += zv * wh[i];
      sc += zv * wc[i];
    }
    if (l == 0) { h0a[m * HPITCH + n] = (bf16_t)sh; c0[m * HPITCH + n] = sc; }
    else        { h1a[m * HPITCH + n] = (bf16_t)sh; c1[m * HPITCH + n] = sc; }
  }
  __syncthreads();

  const v8f vzero = {0, 0, 0, 0, 0, 0, 0, 0};

#pragma unroll 1
  for (int t = 0; t < SLEN; ++t) {
    bf16_t* h0c = (t & 1) ? h0b : h0a;
    bf16_t* h0n = (t & 1) ? h0a : h0b;
    bf16_t* h1c = (t & 1) ? h1b : h1a;
    bf16_t* h1n = (t & 1) ? h1a : h1b;

    // ---- layer 0: gates = (b_ih0+b_hh0) + h0 @ W_hh0^T  (input is zeros) --
#pragma unroll 1
    for (int hb = 0; hb < HTPW; ++hb) {
      int ht = wave * HTPW + hb;                 // h-tile 0..63
      v8f acc[4] = {vzero, vzero, vzero, vzero};
#pragma unroll 4
      for (int kc = 0; kc < KCH; ++kc) {
        v16bf a = load_afrag_lds(h0c, kc * 32, lane);
#pragma unroll
        for (int g = 0; g < 4; ++g) {
          v16bf b = load_bfrag(Bhh0 + ((long long)((g * 64 + ht) * KCH + kc)) * 512, lane);
          acc[g] = wmma_bf16(a, b, acc[g]);
        }
      }
      cell_update(acc, ht, lane, c0, h0n, b_ih0, b_hh0);
    }
    __syncthreads();

    // ---- layer 1: gates = b + h1_l0[t] @ W_ih1^T + h1 @ W_hh1^T -----------
#pragma unroll 1
    for (int hb = 0; hb < HTPW; ++hb) {
      int ht = wave * HTPW + hb;
      v8f acc[4] = {vzero, vzero, vzero, vzero};
#pragma unroll 4
      for (int kc = 0; kc < KCH; ++kc) {        // x-projection from layer 0
        v16bf a = load_afrag_lds(h0n, kc * 32, lane);
#pragma unroll
        for (int g = 0; g < 4; ++g) {
          v16bf b = load_bfrag(Bih1 + ((long long)((g * 64 + ht) * KCH + kc)) * 512, lane);
          acc[g] = wmma_bf16(a, b, acc[g]);
        }
      }
#pragma unroll 4
      for (int kc = 0; kc < KCH; ++kc) {        // recurrent term
        v16bf a = load_afrag_lds(h1c, kc * 32, lane);
#pragma unroll
        for (int g = 0; g < 4; ++g) {
          v16bf b = load_bfrag(Bhh1 + ((long long)((g * 64 + ht) * KCH + kc)) * 512, lane);
          acc[g] = wmma_bf16(a, b, acc[g]);
        }
      }
      cell_update(acc, ht, lane, c1, h1n, b_ih1, b_hh1);
    }
    __syncthreads();

    // ---- output head: sigmoid(h2 @ W_out^T + b_out), 49 cols (pad 64) ----
    if (wave < 4) {
      v8f acc = vzero;
#pragma unroll 4
      for (int kc = 0; kc < KCH; ++kc) {
        v16bf a = load_afrag_lds(h1n, kc * 32, lane);
        v16bf b = load_bfrag(Bout + ((long long)(wave * KCH + kc)) * 512, lane);
        acc = wmma_bf16(a, b, acc);
      }
      int o = wave * 16 + (lane & 15);
      if (o < NOUT) {
        float bo = b_out[o];
        int half = lane >> 4;
#pragma unroll
        for (int v = 0; v < 8; ++v) {
          int m = v + half * 8;
          long long b_ = bbase + m;
          out[(b_ * SLEN + t) * NOUT + o] = sigmoidf_(acc[v] + bo);
        }
      }
    }
    __syncthreads();
  }
}

// ---------------------------------------------------------------------------
extern "C" void kernel_launch(void* const* d_in, const int* in_sizes, int n_in,
                              void* d_out, int out_size, void* d_ws, size_t ws_size,
                              hipStream_t stream) {
  (void)in_sizes; (void)n_in; (void)out_size; (void)ws_size;
  const float* z      = (const float*)d_in[0];
  const float* W_hid  = (const float*)d_in[1];
  const float* b_hid  = (const float*)d_in[2];
  const float* W_cell = (const float*)d_in[3];
  const float* b_cell = (const float*)d_in[4];
  /* d_in[5] = W_ih0: multiplied by zeros in the reference — unused */
  const float* W_hh0  = (const float*)d_in[6];
  const float* b_ih0  = (const float*)d_in[7];
  const float* b_hh0  = (const float*)d_in[8];
  const float* W_ih1  = (const float*)d_in[9];
  const float* W_hh1  = (const float*)d_in[10];
  const float* b_ih1  = (const float*)d_in[11];
  const float* b_hh1  = (const float*)d_in[12];
  const float* W_out  = (const float*)d_in[13];
  const float* b_out  = (const float*)d_in[14];
  float* out = (float*)d_out;

  // Workspace: bf16 pre-swizzled weights (~24.3 MB total)
  bf16_t* Bhh0 = (bf16_t*)d_ws;
  bf16_t* Bih1 = Bhh0 + 4096LL * 1024;
  bf16_t* Bhh1 = Bih1 + 4096LL * 1024;
  bf16_t* Bout = Bhh1 + 4096LL * 1024;

  {
    const int thr = 256;
    long long tot = 4096LL * 1024;
    int blocks = (int)((tot + thr - 1) / thr);
    pack_weights<<<blocks, thr, 0, stream>>>(W_hh0, Bhh0, 256, 4096, KCH, HDIM);
    pack_weights<<<blocks, thr, 0, stream>>>(W_ih1, Bih1, 256, 4096, KCH, HDIM);
    pack_weights<<<blocks, thr, 0, stream>>>(W_hh1, Bhh1, 256, 4096, KCH, HDIM);
    long long tot2 = 4LL * KCH * 512;
    int blocks2 = (int)((tot2 + thr - 1) / thr);
    pack_weights<<<blocks2, thr, 0, stream>>>(W_out, Bout, 4, NOUT, KCH, HDIM);
  }

  const int HB = BT * HPITCH * (int)sizeof(bf16_t);
  const int CB = BT * HPITCH * (int)sizeof(float);
  const int smem_bytes = 4 * HB + 2 * CB;   // 264,704 B < 320 KB WGP LDS
  (void)hipFuncSetAttribute((const void*)lstm_fused,
                            hipFuncAttributeMaxDynamicSharedMemorySize,
                            smem_bytes);

  lstm_fused<<<512 / BT, NTHREADS, smem_bytes, stream>>>(
      z, W_hid, b_hid, W_cell, b_cell,
      b_ih0, b_hh0, b_ih1, b_hh1, b_out,
      Bhh0, Bih1, Bhh1, Bout, out);
}
